// PryIO_38560216383637
// MI455X (gfx1250) — matrix operand order repaired and checked by reading, not compiled, not run on hardware
//
#include <hip/hip_runtime.h>
#include <hip/hip_bf16.h>
#include <stdint.h>

// ---------------------------------------------------------------------------
// N=8 T=256 B=4 D=512 V=32000; rows = N*T*B = 8192
// GEMM [8192 x 512] x [512 x 32000] with fused argmax over V.
// bf16 WMMA (V_WMMA_F32_16X16X32_BF16), f32 accumulate. Compute bound
// (268 GFLOP vs ~41 MB HBM traffic; W fits in 192 MB L2).
// ---------------------------------------------------------------------------

#define N_    8
#define T_    256
#define B_    4
#define D_    512
#define V_    32000
#define ROWS_ 8192
#define VITER 125             // 32000 / (8 waves * 2 tiles * 16 cols)
#define APAD  520             // A-row stride in LDS (bf16 units): 1040 B,
                              // makes each 16-lane half bank-conflict free

typedef __attribute__((ext_vector_type(16))) __bf16 v16bf;
typedef __attribute__((ext_vector_type(8)))  float  v8f;
typedef unsigned int v4u __attribute__((ext_vector_type(4)));
typedef int          v8i __attribute__((ext_vector_type(8)));
typedef int          v4i __attribute__((ext_vector_type(4)));

union AFrag { v16bf v; uint4 q[2]; };

#if defined(__has_builtin)
#  if __has_builtin(__builtin_amdgcn_tensor_load_to_lds) && \
      __has_builtin(__builtin_amdgcn_s_wait_tensorcnt)
#    define HAVE_TDM 1
#  endif
#endif
#ifndef HAVE_TDM
#  define HAVE_TDM 0
#endif

// f32 -> bf16 RNE
__device__ __forceinline__ unsigned short f2bf(float f) {
    unsigned int u = __builtin_bit_cast(unsigned int, f);
    unsigned int r = u + 0x7FFFu + ((u >> 16) & 1u);
    return (unsigned short)(r >> 16);
}

// ---------------------------------------------------------------------------
// Kernel 0: f32 -> bf16 conversion
// ---------------------------------------------------------------------------
__global__ void convert_bf16_kernel(const float* __restrict__ src,
                                    unsigned short* __restrict__ dst,
                                    int n4) {
    int idx = blockIdx.x * blockDim.x + threadIdx.x;
    int stride = gridDim.x * blockDim.x;
    const float4* s4 = (const float4*)src;
    uint2* d2 = (uint2*)dst;
    for (int i = idx; i < n4; i += stride) {
        float4 f = s4[i];
        uint2 o;
        o.x = (unsigned)f2bf(f.x) | ((unsigned)f2bf(f.y) << 16);
        o.y = (unsigned)f2bf(f.z) | ((unsigned)f2bf(f.w) << 16);
        d2[i] = o;
    }
}

// ---------------------------------------------------------------------------
// Kernel 1: fused GEMM + argmax. 512 blocks x 256 threads (8 waves).
// Each wave: two 16-col N tiles sharing one A fragment -> 2 wmma per
// A-load, 2 independent accumulators; K pipelined with rotating frags.
// ---------------------------------------------------------------------------
__device__ __forceinline__ void loadB(AFrag& b, const unsigned short* wrow, int kb) {
    b.q[0] = *(const uint4*)(wrow + kb);
    b.q[1] = *(const uint4*)(wrow + kb + 8);
}
__device__ __forceinline__ void loadA(AFrag& a, const unsigned short* arow, int kb) {
    a.q[0] = *(const uint4*)(arow + kb);
    a.q[1] = *(const uint4*)(arow + kb + 8);
}

__global__ void __launch_bounds__(256)
gemm_argmax_kernel(const unsigned short* __restrict__ Ab,   // [8192 x 512] bf16
                   const unsigned short* __restrict__ Wb,   // [32000 x 512] bf16
                   int* __restrict__ maxidx)                // [8192]
{
    __shared__ unsigned short sA[16 * APAD];  // 16,640 B (padded rows)
    __shared__ float sVal[16 * 128];          // 8 KB
    __shared__ int   sIdx[16 * 128];          // 8 KB

    const int tid  = threadIdx.x;
    const int wave = tid >> 5;
    const int lane = tid & 31;
    const int nIdx = lane & 15;
    const int g    = lane >> 4;
    const int row0 = blockIdx.x * 16;

    // --- Stage A tile (16 x 512 bf16, contiguous 16 KB in global) into LDS
    //     with 16 B/row padding.  TDM path: one tensor_load_to_lds issued by
    //     wave 0; D# = 2D tile 128x16 of 8 B units, pad 4 DWORDs per 256. ---
#if HAVE_TDM
    if (tid < 32) {
        unsigned lds = (unsigned)(uintptr_t)(&sA[0]);
        unsigned long long ga = (unsigned long long)(uintptr_t)(Ab + (size_t)row0 * D_);
        // group0: count=1 | lds_addr | global_addr[56:0] | type=2
        v4u g0 = { 1u, lds, (unsigned)(ga & 0xffffffffu),
                   (unsigned)((ga >> 32) & 0x01ffffffu) | (2u << 30) };
        // group1: data_size=3(8B), pad_enable, pad_interval=7(256 dw=1024B),
        // pad_amount=3(4 dw=16B); tensor_dim0=128, tensor_dim1=16,
        // tile_dim0=128, tile_dim1=16, stride0=128
        v8i g1 = { (int)0x07D30000u,       // mask=0 | ds=3 | pad fields
                   (int)0x00800000u,       // tensor_dim0[15:0]=128 << 16
                   (int)0x00100000u,       // tensor_dim1[15:0]=16  << 16
                   (int)0x00800000u,       // tile_dim0=128 << 16
                   16,                     // tile_dim1=16
                   128,                    // tensor_dim0_stride=128
                   0, 0 };
        v4i z4 = {0, 0, 0, 0};
#if __clang_major__ >= 23
        v8i z8 = {0, 0, 0, 0, 0, 0, 0, 0};
        __builtin_amdgcn_tensor_load_to_lds(g0, g1, z4, z4, z8, 0);
#else
        __builtin_amdgcn_tensor_load_to_lds(g0, g1, z4, z4, 0);
#endif
        __builtin_amdgcn_s_wait_tensorcnt(0);
    }
#else
    {
        const uint4* gsrc = (const uint4*)(Ab + (size_t)row0 * D_);
        #pragma unroll
        for (int j = 0; j < 4; ++j) {
            int i = tid + j * 256;            // 1024 uint4 total, 64/row
            int r = i >> 6, c = i & 63;
            *(uint4*)(sA + r * APAD + c * 8) = gsrc[i];
        }
    }
#endif
    __syncthreads();

    const unsigned short* arow = sA + nIdx * APAD + g * 16;

    float bestv0[8], bestv1[8];
    int   besti0[8], besti1[8];
    #pragma unroll
    for (int r = 0; r < 8; ++r) {
        bestv0[r] = -3.4e38f; besti0[r] = 0;
        bestv1[r] = -3.4e38f; besti1[r] = 0;
    }

    for (int vloop = 0; vloop < VITER; ++vloop) {
        const int vbase = vloop * 256 + wave * 32;
        const unsigned short* wrow0 = Wb + (size_t)(vbase + nIdx) * D_ + g * 16;
        const unsigned short* wrow1 = wrow0 + (size_t)16 * D_;

        v8f c0 = {}, c1 = {};
        AFrag a, b0, b1;
        loadA(a, arow, 0); loadB(b0, wrow0, 0); loadB(b1, wrow1, 0);

        #pragma unroll
        for (int kb = 32; kb <= D_; kb += 32) {
            AFrag an, b0n, b1n;
            if (kb < D_) {                     // prefetch next K-step frags
                loadA(an, arow, kb);
                loadB(b0n, wrow0, kb);
                loadB(b1n, wrow1, kb);
            }
            c0 = __builtin_amdgcn_wmma_f32_16x16x32_bf16(
                     false, a.v, false, b0.v, (short)0, c0, false, false);
            c1 = __builtin_amdgcn_wmma_f32_16x16x32_bf16(
                     false, a.v, false, b1.v, (short)0, c1, false, false);
            if (kb < D_) { a = an; b0 = b0n; b1 = b1n; }
        }

        if (vloop + 1 < VITER) {   // global_prefetch_b8 for next V tile
            __builtin_prefetch(Wb + (size_t)(vbase + 256 + nIdx) * D_, 0, 1);
            __builtin_prefetch(Wb + (size_t)(vbase + 272 + nIdx) * D_, 0, 1);
        }

        const int v0 = vbase + nIdx, v1 = v0 + 16;
        #pragma unroll
        for (int r = 0; r < 8; ++r) {
            float cv0 = c0[r], cv1 = c1[r];
            if (cv0 > bestv0[r]) { bestv0[r] = cv0; besti0[r] = v0; }
            if (cv1 > bestv1[r]) { bestv1[r] = cv1; besti1[r] = v1; }
        }
    }

    // merge the two tiles in-register (argmax, lowest index on tie)
    #pragma unroll
    for (int r = 0; r < 8; ++r) {
        if (bestv1[r] > bestv0[r] ||
            (bestv1[r] == bestv0[r] && besti1[r] < besti0[r])) {
            bestv0[r] = bestv1[r]; besti0[r] = besti1[r];
        }
    }

    {
        const int slot = wave * 16 + nIdx;
        #pragma unroll
        for (int r = 0; r < 8; ++r) {
            const int m = g * 8 + r;
            sVal[m * 128 + slot] = bestv0[r];
            sIdx[m * 128 + slot] = besti0[r];
        }
    }
    __syncthreads();

    if (tid < 16) {
        float bv = -3.4e38f;
        int   bi = 0x7fffffff;
        for (int s = 0; s < 128; ++s) {
            float v = sVal[tid * 128 + s];
            int   i = sIdx[tid * 128 + s];
            if (v > bv || (v == bv && i < bi)) { bv = v; bi = i; }
        }
        maxidx[row0 + tid] = bi;
    }
}

// ---------------------------------------------------------------------------
// Kernel 2: scores[row] = sqrt(D) * dot(outputs[row/B], W[maxidx[row]])
// ---------------------------------------------------------------------------
__global__ void scores_kernel(const float* __restrict__ outputs,
                              const float* __restrict__ W,
                              const int* __restrict__ maxidx,
                              float* __restrict__ scores)
{
    __shared__ float red[128];
    const int row = blockIdx.x;
    const int tid = threadIdx.x;
    const float* o = outputs + (size_t)(row >> 2) * D_;
    const float* w = W + (size_t)maxidx[row] * D_;
    float p = 0.f;
    #pragma unroll
    for (int d = tid; d < D_; d += 128) p += o[d] * w[d];
    red[tid] = p;
    __syncthreads();
    for (int s = 64; s > 0; s >>= 1) {
        if (tid < s) red[tid] += red[tid + s];
        __syncthreads();
    }
    if (tid == 0) scores[row] = red[0] * 22.62741699796952f;  // sqrt(512)
}

// ---------------------------------------------------------------------------
// Kernel 3: zero the output mask
// ---------------------------------------------------------------------------
__global__ void zero_kernel(float* __restrict__ p, int n) {
    int i = blockIdx.x * blockDim.x + threadIdx.x;
    if (i < n) p[i] = 0.f;
}

// ---------------------------------------------------------------------------
// Kernel 4: sequential scan, Threefry-2x32 (jax fold_in + bernoulli bits)
// ---------------------------------------------------------------------------
__device__ __forceinline__ unsigned rotl32(unsigned x, int r) {
    return (x << r) | (x >> (32 - r));
}
__device__ __forceinline__ void tf2x32(unsigned k0, unsigned k1,
                                       unsigned c0, unsigned c1,
                                       unsigned& o0, unsigned& o1) {
    const unsigned ks2 = 0x1BD11BDAu ^ k0 ^ k1;
    unsigned x0 = c0 + k0, x1 = c1 + k1;
    auto R4 = [&](int r0, int r1, int r2, int r3) {
        x0 += x1; x1 = rotl32(x1, r0); x1 ^= x0;
        x0 += x1; x1 = rotl32(x1, r1); x1 ^= x0;
        x0 += x1; x1 = rotl32(x1, r2); x1 ^= x0;
        x0 += x1; x1 = rotl32(x1, r3); x1 ^= x0;
    };
    R4(13, 15, 26, 6);   x0 += k1;  x1 += ks2 + 1u;
    R4(17, 29, 16, 24);  x0 += ks2; x1 += k0 + 2u;
    R4(13, 15, 26, 6);   x0 += k0;  x1 += k1 + 3u;
    R4(17, 29, 16, 24);  x0 += k1;  x1 += ks2 + 4u;
    R4(13, 15, 26, 6);   x0 += ks2; x1 += k0 + 5u;
    o0 = x0; o1 = x1;
}

__global__ void scan_kernel(const float* __restrict__ scores,
                            float* __restrict__ mask)
{
    const int n = threadIdx.x;
    if (n >= N_) return;
    int x = 0, y = 0;
    for (int t = 0; t < T_; ++t) {
        unsigned fk0, fk1;
        tf2x32(0u, 42u, 0u, (unsigned)t, fk0, fk1);       // fold_in(key(42), t)
        unsigned b0, b1, bits;
        if (n < 4) { tf2x32(fk0, fk1, (unsigned)n, (unsigned)(n + 4), b0, b1); bits = b0; }
        else       { tf2x32(fk0, fk1, (unsigned)(n - 4), (unsigned)n, b0, b1); bits = b1; }
        float u = __builtin_bit_cast(float, (bits >> 9) | 0x3f800000u) - 1.0f;

        const int idx0 = x * B_ + y;
        const float scx0 = scores[n * (T_ * B_) + idx0];
        const float scx1 = scores[n * (T_ * B_) + t * B_];
        const float prob = 1.0f / (1.0f + __expf(-(scx0 - scx1) * 0.1f));

        mask[n * (T_ * B_) + idx0] = 1.0f;

        int accept = (u < prob) ? 1 : 0;
        const int y1 = y + accept;
        accept *= (y1 < B_) ? 1 : 0;
        y = (y + accept) * accept;
        x = x * accept + (t + 1) * (1 - accept);
    }
}

// ---------------------------------------------------------------------------
// Workspace layout (needs ~42 MB):
//   [0)         Wb bf16 32000*512 = 32,768,000 B
//   [32768000)  Ab bf16  8192*512 =  8,388,608 B
//   [41156608)  maxidx int 8192
//   [41189376)  scores f32 8192
// ---------------------------------------------------------------------------
extern "C" void kernel_launch(void* const* d_in, const int* in_sizes, int n_in,
                              void* d_out, int out_size, void* d_ws, size_t ws_size,
                              hipStream_t stream) {
    const float* outputs       = (const float*)d_in[0];
    const float* block_outputs = (const float*)d_in[1];
    const float* W             = (const float*)d_in[2];
    float* mask = (float*)d_out;

    char* ws = (char*)d_ws;
    unsigned short* Wb   = (unsigned short*)(ws);
    unsigned short* Ab   = (unsigned short*)(ws + 32768000u);
    int*            midx = (int*)           (ws + 41156608u);
    float*          sco  = (float*)         (ws + 41189376u);

    convert_bf16_kernel<<<4000, 256, 0, stream>>>(W, Wb, (V_ * D_) / 4);
    convert_bf16_kernel<<<1024, 256, 0, stream>>>(block_outputs, Ab, (ROWS_ * D_) / 4);

    gemm_argmax_kernel<<<ROWS_ / 16, 256, 0, stream>>>(Ab, Wb, midx);

    scores_kernel<<<ROWS_, 128, 0, stream>>>(outputs, W, midx, sco);

    zero_kernel<<<(ROWS_ + 255) / 256, 256, 0, stream>>>(mask, ROWS_);
    scan_kernel<<<1, 32, 0, stream>>>(sco, mask);
}
